// MLA_59227599012426
// MI455X (gfx1250) — matrix operand order repaired and checked by reading, not compiled, not run on hardware
//
#include <hip/hip_runtime.h>
#include <hip/hip_bf16.h>
#include <cstdint>

#define S_LEN 2048
#define DIMN  2048
#define NHEAD 16
#define QLORA 512
#define KVLORA 512
#define NOPE  128
#define ROPE_D 64
#define VDIM  128
#define IHEAD 16
#define IDIM  128
#define TOPK  512
#define QKDIM 192   // NOPE + ROPE
#define BIGNEG (-1e30f)

typedef __bf16 bf16_t;
typedef bf16_t bf16x8  __attribute__((ext_vector_type(8)));
typedef bf16_t bf16x16 __attribute__((ext_vector_type(16)));
typedef float  floatx8 __attribute__((ext_vector_type(8)));

typedef __attribute__((address_space(3))) bf16_t* lds_bf16p;

union Frag16 { bf16x16 v; bf16x8 h[2]; };

__device__ __forceinline__ floatx8 fzero8() {
  floatx8 v;
  #pragma unroll
  for (int i = 0; i < 8; ++i) v[i] = 0.0f;
  return v;
}

__device__ __forceinline__ floatx8 wmma_bf16(const Frag16& a, const Frag16& b, floatx8 c) {
  return __builtin_amdgcn_wmma_f32_16x16x32_bf16(false, a.v, false, b.v, (short)0, c, false, false);
}

// Async copy of 16 bytes global -> LDS (per lane), tracked by ASYNCcnt.
__device__ __forceinline__ void async_copy_b128(const bf16_t* g, bf16_t* l) {
  lds_bf16p lp = (lds_bf16p)l;
  asm volatile("global_load_async_to_lds_b128 %0, %1, off" :: "v"(lp), "v"(g) : "memory");
}
__device__ __forceinline__ void async_wait0() {
  asm volatile("s_wait_asynccnt 0x0" ::: "memory");
}

// ---------------------------------------------------------------------------
// Generic GEMM: C[M,N] = A[M,K] (f32 row-major) * W[N,K]^T (f32 row-major)
// Requires M%128==0, N%64==0, K%32==0.
// ---------------------------------------------------------------------------
#define GBM 128
#define GBN 64
#define GBK 32
#define GLD 48  // padded LDS row stride (bf16 elems), 96B (16B aligned)

__global__ __launch_bounds__(256) void mla_gemm_bt(const float* __restrict__ A,
                                                   const float* __restrict__ W,
                                                   float* __restrict__ C,
                                                   int M, int N, int K) {
  __shared__ bf16_t la[GBM * GLD];
  __shared__ bf16_t lb[GBN * GLD];
  const int tid   = threadIdx.x;
  const int lane  = tid & 31;
  const int wave  = tid >> 5;
  const int wm    = (wave & 3) * 32;   // 4 waves along M
  const int wn    = (wave >> 2) * 32;  // 2 waves along N
  const int m0    = blockIdx.x * GBM;
  const int n0    = blockIdx.y * GBN;
  const int khalf = lane >> 4;
  const int l15   = lane & 15;
  (void)M;

  floatx8 acc[2][2];
  #pragma unroll
  for (int a = 0; a < 2; ++a)
    #pragma unroll
    for (int b = 0; b < 2; ++b) acc[a][b] = fzero8();

  for (int k0 = 0; k0 < K; k0 += GBK) {
    for (int i = tid; i < GBM * GBK; i += 256) {
      int r = i >> 5, c = i & 31;
      la[r * GLD + c] = (bf16_t)A[(size_t)(m0 + r) * K + k0 + c];
    }
    for (int i = tid; i < GBN * GBK; i += 256) {
      int r = i >> 5, c = i & 31;
      lb[r * GLD + c] = (bf16_t)W[(size_t)(n0 + r) * K + k0 + c];
    }
    // prefetch next K tiles into cache while this tile is consumed
    if (k0 + GBK < K) {
      __builtin_prefetch(&A[(size_t)(m0 + (tid & 127)) * K + k0 + GBK], 0, 1);
      __builtin_prefetch(&W[(size_t)(n0 + (tid & 63)) * K + k0 + GBK], 0, 1);
    }
    __syncthreads();

    Frag16 af[2], bfrag[2];
    #pragma unroll
    for (int a = 0; a < 2; ++a) {
      const bf16_t* p = &la[(wm + a * 16 + l15) * GLD + 8 * khalf];
      af[a].h[0] = *(const bf16x8*)p;
      af[a].h[1] = *(const bf16x8*)(p + 16);
    }
    #pragma unroll
    for (int b = 0; b < 2; ++b) {
      const bf16_t* p = &lb[(wn + b * 16 + l15) * GLD + 8 * khalf];
      bfrag[b].h[0] = *(const bf16x8*)p;
      bfrag[b].h[1] = *(const bf16x8*)(p + 16);
    }
    #pragma unroll
    for (int a = 0; a < 2; ++a)
      #pragma unroll
      for (int b = 0; b < 2; ++b)
        acc[a][b] = wmma_bf16(af[a], bfrag[b], acc[a][b]);
    __syncthreads();
  }

  #pragma unroll
  for (int a = 0; a < 2; ++a)
    #pragma unroll
    for (int b = 0; b < 2; ++b)
      #pragma unroll
      for (int j = 0; j < 8; ++j) {
        int m = m0 + wm + a * 16 + j + 8 * khalf;
        int n = n0 + wn + b * 16 + l15;
        C[(size_t)m * N + n] = acc[a][b][j];
      }
}

// ---------------------------------------------------------------------------
// RoPE tables: cos/sin [S, 32]
// ---------------------------------------------------------------------------
__global__ void mla_rope_tables(float* cosr, float* sinr, int total) {
  int idx = blockIdx.x * blockDim.x + threadIdx.x;
  if (idx >= total) return;
  int pos = idx >> 5, i = idx & 31;
  float freq = powf(10000.0f, -((float)i) / 32.0f);
  float ang = (float)pos * freq;
  cosr[idx] = cosf(ang);
  sinr[idx] = sinf(ang);
}

// Apply interleaved-pair RoPE in place. total = S*nh*32
__global__ void mla_rope_apply(float* t, const float* cosr, const float* sinr,
                               int nh, int rowStride, int headStride, int peOff, int total) {
  int idx = blockIdx.x * blockDim.x + threadIdx.x;
  if (idx >= total) return;
  int i = idx & 31;
  int sh = idx >> 5;
  int h = sh % nh;
  int s = sh / nh;
  float* p = t + (size_t)s * rowStride + (size_t)h * headStride + peOff + 2 * i;
  float c = cosr[s * 32 + i], sn = sinr[s * 32 + i];
  float x0 = p[0], x1 = p[1];
  p[0] = x0 * c - x1 * sn;
  p[1] = x0 * sn + x1 * c;
}

// ---------------------------------------------------------------------------
// RMSNorm rows (src may be strided wider than n); dst may alias src.
// ---------------------------------------------------------------------------
__global__ __launch_bounds__(256) void mla_rmsnorm(const float* __restrict__ src, int sstride,
                                                   float* __restrict__ dst, int dstride,
                                                   const float* __restrict__ w, int n) {
  __shared__ float red[256];
  int row = blockIdx.x, tid = threadIdx.x;
  const float* s = src + (size_t)row * sstride;
  float vals[4];
  int cnt = 0;
  float ss = 0.f;
  for (int i = tid; i < n; i += 256) { float v = s[i]; vals[cnt++] = v; ss += v * v; }
  red[tid] = ss;
  __syncthreads();
  for (int o = 128; o > 0; o >>= 1) {
    if (tid < o) red[tid] += red[tid + o];
    __syncthreads();
  }
  float inv = rsqrtf(red[0] / (float)n + 1e-6f);
  float* d = dst + (size_t)row * dstride;
  cnt = 0;
  for (int i = tid; i < n; i += 256) d[i] = w[i] * vals[cnt++] * inv;
}

// LayerNorm in place, n == blockDim.x == 128
__global__ __launch_bounds__(128) void mla_layernorm(float* __restrict__ t,
                                                     const float* __restrict__ w,
                                                     const float* __restrict__ b, int n) {
  __shared__ float rs[128], rs2[128];
  int row = blockIdx.x, tid = threadIdx.x;
  float* p = t + (size_t)row * n;
  float v = p[tid];
  rs[tid] = v;
  rs2[tid] = v * v;
  __syncthreads();
  for (int o = 64; o > 0; o >>= 1) {
    if (tid < o) { rs[tid] += rs[tid + o]; rs2[tid] += rs2[tid + o]; }
    __syncthreads();
  }
  float mu = rs[0] / (float)n;
  float var = rs2[0] / (float)n - mu * mu;
  float inv = rsqrtf(var + 1e-6f);
  p[tid] = w[tid] * ((v - mu) * inv) + b[tid];
}

// ---------------------------------------------------------------------------
// f32 -> bf16 pack
// ---------------------------------------------------------------------------
__global__ void mla_f2bf(const float* __restrict__ src, bf16_t* __restrict__ dst, int n) {
  int idx = blockIdx.x * blockDim.x + threadIdx.x;
  if (idx < n) dst[idx] = (bf16_t)src[idx];
}

// Build K (nope from kvb, roped pe from kv) bf16 buffer: [key][h][192]
__global__ void mla_build_k(const float* __restrict__ kvb, const float* __restrict__ kv,
                            bf16_t* __restrict__ k_bf, int total) {
  int idx = blockIdx.x * blockDim.x + threadIdx.x;
  if (idx >= total) return;
  int d = idx % QKDIM;
  int sh = idx / QKDIM;
  int h = sh % NHEAD;
  int s = sh / NHEAD;
  float v;
  if (d < NOPE) v = kvb[((size_t)s * NHEAD + h) * (NOPE + VDIM) + d];
  else          v = kv[(size_t)s * (KVLORA + ROPE_D) + KVLORA + (d - NOPE)];
  k_bf[idx] = (bf16_t)v;
}

// Build V in head-major TRANSPOSED layout: vt[h][d][key]  (key contiguous)
// so the flash kernel's LDS tile is a straight async copy.
__global__ void mla_build_vt(const float* __restrict__ kvb, bf16_t* __restrict__ vt, int total) {
  int idx = blockIdx.x * blockDim.x + threadIdx.x;
  if (idx >= total) return;
  int s = idx % S_LEN;
  int hd = idx / S_LEN;         // hd = h*VDIM + d
  int d = hd % VDIM;
  int h = hd / VDIM;
  vt[idx] = (bf16_t)kvb[((size_t)s * NHEAD + h) * (NOPE + VDIM) + NOPE + d];
}

// ---------------------------------------------------------------------------
// FWHT-128 (Sylvester Hadamard, symmetric) + 128^-0.5 scale, emit bf16.
// ---------------------------------------------------------------------------
__global__ __launch_bounds__(128) void mla_hadamard(const float* __restrict__ src,
                                                    bf16_t* __restrict__ dst) {
  __shared__ float v[128];
  int vec = blockIdx.x, tid = threadIdx.x;
  v[tid] = src[(size_t)vec * 128 + tid];
  __syncthreads();
  #pragma unroll
  for (int h = 1; h < 128; h <<= 1) {
    float a = v[tid], b = v[tid ^ h];
    __syncthreads();
    v[tid] = (tid & h) ? (b - a) : (a + b);
    __syncthreads();
  }
  dst[(size_t)vec * 128 + tid] = (bf16_t)(v[tid] * 0.08838834764831845f);
}

// ---------------------------------------------------------------------------
// w[s,h] = (x[s,:] . idx_wproj[h,:]) * IH^-0.5 * ID^-0.5
// ---------------------------------------------------------------------------
__global__ void mla_wproj(const float* __restrict__ x, const float* __restrict__ wp,
                          float* __restrict__ w, int total) {
  int idx = blockIdx.x * blockDim.x + threadIdx.x;
  if (idx >= total) return;
  int h = idx % IHEAD;
  int s = idx / IHEAD;
  const float* xr = x + (size_t)s * DIMN;
  const float* wr = wp + (size_t)h * DIMN;
  float acc = 0.f;
  for (int d = 0; d < DIMN; ++d) acc += xr[d] * wr[d];
  w[idx] = acc * 0.022097086912079608f;  // 1/4 * 1/sqrt(128)
}

// ---------------------------------------------------------------------------
// Indexer score: iscore[q,t] = sum_h relu(iq[q,h,:].ik[t,:]) * w[q,h] + causal
// ---------------------------------------------------------------------------
__global__ __launch_bounds__(128) void mla_idx_score(const bf16_t* __restrict__ iq_bf,
                                                     const bf16_t* __restrict__ ik_bf,
                                                     const float* __restrict__ wbuf,
                                                     float* __restrict__ iscore) {
  const int tid = threadIdx.x, lane = tid & 31, wave = tid >> 5;
  const int q0 = blockIdx.x * 16;
  const int t0 = blockIdx.y * 64 + wave * 16;
  const int khalf = lane >> 4, l15 = lane & 15;

  floatx8 acc = fzero8();
  const bf16_t* bBase = ik_bf + (size_t)(t0 + l15) * IDIM;
  for (int h = 0; h < IHEAD; ++h) {
    floatx8 s = fzero8();
    const bf16_t* aBase = iq_bf + ((size_t)(q0 + l15) * IHEAD + h) * IDIM;
    #pragma unroll
    for (int kb = 0; kb < 4; ++kb) {
      Frag16 a, b;
      const bf16_t* ap = aBase + kb * 32 + 8 * khalf;
      a.h[0] = *(const bf16x8*)ap;
      a.h[1] = *(const bf16x8*)(ap + 16);
      const bf16_t* bp = bBase + kb * 32 + 8 * khalf;
      b.h[0] = *(const bf16x8*)bp;
      b.h[1] = *(const bf16x8*)(bp + 16);
      s = wmma_bf16(a, b, s);
    }
    #pragma unroll
    for (int j = 0; j < 8; ++j) {
      int q = q0 + j + 8 * khalf;
      float wv = wbuf[(size_t)q * IHEAD + h];
      float sv = s[j];
      acc[j] += (sv > 0.f ? sv : 0.f) * wv;
    }
  }
  #pragma unroll
  for (int j = 0; j < 8; ++j) {
    int q = q0 + j + 8 * khalf;
    int t = t0 + l15;
    float v = acc[j];
    if (t > q) v += BIGNEG;
    iscore[(size_t)q * S_LEN + t] = v;
  }
}

// ---------------------------------------------------------------------------
// Per-row k-th largest via MSB-first 8-bit radix select.
// ---------------------------------------------------------------------------
__global__ __launch_bounds__(256) void mla_topk_thresh(const float* __restrict__ iscore,
                                                       float* __restrict__ thresh, int k) {
  __shared__ unsigned int hist[256];
  __shared__ unsigned int sh_sel, sh_rem;
  int row = blockIdx.x, tid = threadIdx.x;
  const float* sr = iscore + (size_t)row * S_LEN;
  unsigned int prefix = 0u, pmask = 0u, remaining = (unsigned int)k;
  for (int pass = 3; pass >= 0; --pass) {
    int shift = pass * 8;
    hist[tid] = 0u;
    __syncthreads();
    for (int t = tid; t < S_LEN; t += 256) {
      unsigned int u = __float_as_uint(sr[t]);
      unsigned int key = (u & 0x80000000u) ? ~u : (u | 0x80000000u);
      if ((key & pmask) == prefix) atomicAdd(&hist[(key >> shift) & 255u], 1u);
    }
    __syncthreads();
    if (tid == 0) {
      unsigned int cum = 0, sel = 0, rem = remaining;
      for (int b = 255; b >= 0; --b) {
        unsigned int c = hist[b];
        if (cum + c >= remaining) { sel = (unsigned int)b; rem = remaining - cum; break; }
        cum += c;
      }
      sh_sel = sel;
      sh_rem = rem;
    }
    __syncthreads();
    prefix |= sh_sel << shift;
    pmask |= 0xFFu << shift;
    remaining = sh_rem;
    __syncthreads();
  }
  if (tid == 0) {
    unsigned int key = prefix;
    unsigned int u = (key & 0x80000000u) ? (key & 0x7fffffffu) : ~key;
    thresh[row] = __uint_as_float(u);
  }
}

// ---------------------------------------------------------------------------
// Flash attention with top-k sparse mask. Block: 256 thr (8 waves), one head,
// 128 queries (16 per wave). 32-key chunks staged in LDS via ASYNC copies.
// ---------------------------------------------------------------------------
#define FKEYS 32
__global__ __launch_bounds__(256) void mla_flash_attn(const bf16_t* __restrict__ q_bf,
                                                      const bf16_t* __restrict__ k_bf,
                                                      const bf16_t* __restrict__ vt,
                                                      const float* __restrict__ iscore,
                                                      const float* __restrict__ thresh,
                                                      float* __restrict__ o) {
  __shared__ bf16_t lk[FKEYS * QKDIM];      // [key][192]
  __shared__ bf16_t lvt[VDIM * FKEYS];      // [vdim][key] (already transposed in global)
  __shared__ bf16_t pbuf[8 * 16 * FKEYS];   // per-wave P tiles [m][key]
  const int tid = threadIdx.x, lane = tid & 31, wave = tid >> 5;
  const int h = blockIdx.y;
  const int qtile = blockIdx.x * 128 + wave * 16;
  const int khalf = lane >> 4, l15 = lane & 15;
  const float scale = 0.07216878364870323f;  // (NOPE+ROPE)^-0.5

  Frag16 qf[6];
  {
    const bf16_t* qb = q_bf + ((size_t)(qtile + l15) * NHEAD + h) * QKDIM;
    #pragma unroll
    for (int kb = 0; kb < 6; ++kb) {
      const bf16_t* p = qb + kb * 32 + 8 * khalf;
      qf[kb].h[0] = *(const bf16x8*)p;
      qf[kb].h[1] = *(const bf16x8*)(p + 16);
    }
  }

  floatx8 oacc[8];
  #pragma unroll
  for (int vb = 0; vb < 8; ++vb) oacc[vb] = fzero8();
  float mmax[8], msum[8];
  int qpos[8];
  #pragma unroll
  for (int j = 0; j < 8; ++j) {
    mmax[j] = BIGNEG;
    msum[j] = 0.f;
    qpos[j] = qtile + j + 8 * khalf;
  }

  for (int c = 0; c < S_LEN / FKEYS; ++c) {
    // Async LDS fills (each lane copies 16B; 3+2 per thread, counts divide evenly)
    #pragma unroll
    for (int i = tid; i < FKEYS * (QKDIM / 8); i += 256) {  // 768 chunks of 16B
      int r = i / (QKDIM / 8), cu = i % (QKDIM / 8);
      async_copy_b128(k_bf + ((size_t)(c * FKEYS + r) * NHEAD + h) * QKDIM + cu * 8,
                      &lk[r * QKDIM + cu * 8]);
    }
    #pragma unroll
    for (int i = tid; i < VDIM * (FKEYS / 8); i += 256) {   // 512 chunks of 16B
      int d = i >> 2, ku = i & 3;
      async_copy_b128(vt + ((size_t)h * VDIM + d) * S_LEN + c * FKEYS + ku * 8,
                      &lvt[d * FKEYS + ku * 8]);
    }
    async_wait0();
    __syncthreads();

    // S = Q K^T for both 16-key subtiles
    floatx8 s0 = fzero8(), s1 = fzero8();
    #pragma unroll
    for (int kb = 0; kb < 6; ++kb) {
      Frag16 b0, b1;
      const bf16_t* p0 = &lk[(size_t)l15 * QKDIM + kb * 32 + 8 * khalf];
      b0.h[0] = *(const bf16x8*)p0;
      b0.h[1] = *(const bf16x8*)(p0 + 16);
      const bf16_t* p1 = &lk[(size_t)(16 + l15) * QKDIM + kb * 32 + 8 * khalf];
      b1.h[0] = *(const bf16x8*)p1;
      b1.h[1] = *(const bf16x8*)(p1 + 16);
      s0 = wmma_bf16(qf[kb], b0, s0);
      s1 = wmma_bf16(qf[kb], b1, s1);
    }

    int key0 = c * FKEYS + l15;
    int key1 = key0 + 16;
    #pragma unroll
    for (int j = 0; j < 8; ++j) {
      int qp = qpos[j];
      float v0 = s0[j] * scale;
      float v1 = s1[j] * scale;
      float th = thresh[qp];
      bool ok0 = (key0 <= qp) && (iscore[(size_t)qp * S_LEN + key0] >= th);
      bool ok1 = (key1 <= qp) && (iscore[(size_t)qp * S_LEN + key1] >= th);
      if (!ok0) v0 = BIGNEG;
      if (!ok1) v1 = BIGNEG;
      float rm = fmaxf(v0, v1);
      #pragma unroll
      for (int off = 1; off < 16; off <<= 1) rm = fmaxf(rm, __shfl_xor(rm, off, 32));
      float nm = fmaxf(mmax[j], rm);
      float corr = __expf(mmax[j] - nm);
      float p0 = ok0 ? __expf(v0 - nm) : 0.f;
      float p1 = ok1 ? __expf(v1 - nm) : 0.f;
      float rs = p0 + p1;
      #pragma unroll
      for (int off = 1; off < 16; off <<= 1) rs += __shfl_xor(rs, off, 32);
      msum[j] = msum[j] * corr + rs;
      mmax[j] = nm;
      #pragma unroll
      for (int vb = 0; vb < 8; ++vb) oacc[vb][j] *= corr;
      int m = j + 8 * khalf;
      bf16_t* pb = &pbuf[((size_t)wave * 16 + m) * FKEYS];
      pb[l15] = (bf16_t)p0;
      pb[16 + l15] = (bf16_t)p1;
    }

    // O += P V  (wave-private pbuf; LDS ops are in-order within a wave)
    Frag16 pf;
    {
      const bf16_t* p = &pbuf[((size_t)wave * 16 + l15) * FKEYS + 8 * khalf];
      pf.h[0] = *(const bf16x8*)p;
      pf.h[1] = *(const bf16x8*)(p + 16);
    }
    #pragma unroll
    for (int vb = 0; vb < 8; ++vb) {
      Frag16 bv;
      const bf16_t* p = &lvt[(size_t)(vb * 16 + l15) * FKEYS + 8 * khalf];
      bv.h[0] = *(const bf16x8*)p;
      bv.h[1] = *(const bf16x8*)(p + 16);
      oacc[vb] = wmma_bf16(pf, bv, oacc[vb]);
    }
    __syncthreads();
  }

  #pragma unroll
  for (int j = 0; j < 8; ++j) {
    float inv = (msum[j] > 0.f) ? 1.f / msum[j] : 0.f;
    int m = qtile + j + 8 * khalf;
    #pragma unroll
    for (int vb = 0; vb < 8; ++vb)
      o[((size_t)m * NHEAD + h) * VDIM + vb * 16 + l15] = oacc[vb][j] * inv;
  }
}

// ---------------------------------------------------------------------------
// Host-side orchestration
// ---------------------------------------------------------------------------
extern "C" void kernel_launch(void* const* d_in, const int* in_sizes, int n_in,
                              void* d_out, int out_size, void* d_ws, size_t ws_size,
                              hipStream_t stream) {
  (void)in_sizes; (void)n_in; (void)out_size; (void)ws_size;
  const float* x        = (const float*)d_in[0];
  /* d_in[1] = start_pos (0) ignored */
  const float* wq_a     = (const float*)d_in[2];
  const float* q_norm_w = (const float*)d_in[3];
  const float* wq_b     = (const float*)d_in[4];
  const float* wkv_a    = (const float*)d_in[5];
  const float* kv_norm_w= (const float*)d_in[6];
  const float* wkv_b    = (const float*)d_in[7];
  const float* wo       = (const float*)d_in[8];
  const float* idx_wq_b = (const float*)d_in[9];
  const float* idx_wk   = (const float*)d_in[10];
  const float* idx_kn_w = (const float*)d_in[11];
  const float* idx_kn_b = (const float*)d_in[12];
  const float* idx_wproj= (const float*)d_in[13];
  float* out = (float*)d_out;

  char* ws = (char*)d_ws;
  size_t off = 0;
  auto walloc = [&](size_t bytes) -> void* {
    void* p = ws + off;
    off = (off + bytes + 255) & ~(size_t)255;
    return p;
  };

  float*  cosr   = (float*)walloc((size_t)S_LEN * 32 * 4);
  float*  sinr   = (float*)walloc((size_t)S_LEN * 32 * 4);
  float*  qr     = (float*)walloc((size_t)S_LEN * QLORA * 4);
  float*  q      = (float*)walloc((size_t)S_LEN * NHEAD * QKDIM * 4);
  float*  kv     = (float*)walloc((size_t)S_LEN * (KVLORA + ROPE_D) * 4);
  float*  kvn    = (float*)walloc((size_t)S_LEN * KVLORA * 4);
  float*  kvb    = (float*)walloc((size_t)S_LEN * NHEAD * (NOPE + VDIM) * 4);
  bf16_t* q_bf   = (bf16_t*)walloc((size_t)S_LEN * NHEAD * QKDIM * 2);
  bf16_t* k_bf   = (bf16_t*)walloc((size_t)S_LEN * NHEAD * QKDIM * 2);
  bf16_t* vt_bf  = (bf16_t*)walloc((size_t)NHEAD * VDIM * S_LEN * 2);
  float*  iq     = (float*)walloc((size_t)S_LEN * IHEAD * IDIM * 4);
  bf16_t* iq_bf  = (bf16_t*)walloc((size_t)S_LEN * IHEAD * IDIM * 2);
  float*  ik     = (float*)walloc((size_t)S_LEN * IDIM * 4);
  bf16_t* ik_bf  = (bf16_t*)walloc((size_t)S_LEN * IDIM * 2);
  float*  wbuf   = (float*)walloc((size_t)S_LEN * IHEAD * 4);
  float*  iscore = (float*)walloc((size_t)S_LEN * S_LEN * 4);
  float*  thresh = (float*)walloc((size_t)S_LEN * 4);
  float*  obuf   = (float*)walloc((size_t)S_LEN * NHEAD * VDIM * 4);

  // 1) RoPE tables
  {
    int total = S_LEN * 32;
    mla_rope_tables<<<(total + 255) / 256, 256, 0, stream>>>(cosr, sinr, total);
  }
  // 2) qr = rmsnorm(x @ wq_a^T)
  mla_gemm_bt<<<dim3(S_LEN / GBM, QLORA / GBN), 256, 0, stream>>>(x, wq_a, qr, S_LEN, QLORA, DIMN);
  mla_rmsnorm<<<S_LEN, 256, 0, stream>>>(qr, QLORA, qr, QLORA, q_norm_w, QLORA);
  // 3) q = qr @ wq_b^T ; rope pe part
  mla_gemm_bt<<<dim3(S_LEN / GBM, (NHEAD * QKDIM) / GBN), 256, 0, stream>>>(
      qr, wq_b, q, S_LEN, NHEAD * QKDIM, QLORA);
  {
    int total = S_LEN * NHEAD * 32;
    mla_rope_apply<<<(total + 255) / 256, 256, 0, stream>>>(q, cosr, sinr, NHEAD,
                                                            NHEAD * QKDIM, QKDIM, NOPE, total);
  }
  // 4) kv = x @ wkv_a^T ; rope k_pe in place
  mla_gemm_bt<<<dim3(S_LEN / GBM, (KVLORA + ROPE_D) / GBN), 256, 0, stream>>>(
      x, wkv_a, kv, S_LEN, KVLORA + ROPE_D, DIMN);
  {
    int total = S_LEN * 32;
    mla_rope_apply<<<(total + 255) / 256, 256, 0, stream>>>(kv, cosr, sinr, 1,
                                                            KVLORA + ROPE_D, 0, KVLORA, total);
  }
  // 5) kvn = rmsnorm(kv[:, :512]) ; kvb = kvn @ wkv_b^T
  mla_rmsnorm<<<S_LEN, 256, 0, stream>>>(kv, KVLORA + ROPE_D, kvn, KVLORA, kv_norm_w, KVLORA);
  mla_gemm_bt<<<dim3(S_LEN / GBM, (NHEAD * (NOPE + VDIM)) / GBN), 256, 0, stream>>>(
      kvn, wkv_b, kvb, S_LEN, NHEAD * (NOPE + VDIM), KVLORA);
  // 6) bf16 packs
  {
    int n = S_LEN * NHEAD * QKDIM;
    mla_f2bf<<<(n + 255) / 256, 256, 0, stream>>>(q, q_bf, n);
    mla_build_k<<<(n + 255) / 256, 256, 0, stream>>>(kvb, kv, k_bf, n);
    int nv = NHEAD * VDIM * S_LEN;
    mla_build_vt<<<(nv + 255) / 256, 256, 0, stream>>>(kvb, vt_bf, nv);
  }
  // 7) indexer: iq = qr @ idx_wq_b^T ; ik = layernorm(x @ idx_wk^T)
  mla_gemm_bt<<<dim3(S_LEN / GBM, (IHEAD * IDIM) / GBN), 256, 0, stream>>>(
      qr, idx_wq_b, iq, S_LEN, IHEAD * IDIM, QLORA);
  mla_gemm_bt<<<dim3(S_LEN / GBM, IDIM / GBN), 256, 0, stream>>>(
      x, idx_wk, ik, S_LEN, IDIM, DIMN);
  mla_layernorm<<<S_LEN, 128, 0, stream>>>(ik, idx_kn_w, idx_kn_b, IDIM);
  {
    int total = S_LEN * IHEAD * 32;
    mla_rope_apply<<<(total + 255) / 256, 256, 0, stream>>>(iq, cosr, sinr, IHEAD,
                                                            IHEAD * IDIM, IDIM, 0, total);
    int t2 = S_LEN * 32;
    mla_rope_apply<<<(t2 + 255) / 256, 256, 0, stream>>>(ik, cosr, sinr, 1, IDIM, 0, 0, t2);
  }
  // 8) Hadamard rotate -> bf16
  mla_hadamard<<<S_LEN * IHEAD, 128, 0, stream>>>(iq, iq_bf);
  mla_hadamard<<<S_LEN, 128, 0, stream>>>(ik, ik_bf);
  // 9) head gate w
  {
    int total = S_LEN * IHEAD;
    mla_wproj<<<(total + 255) / 256, 256, 0, stream>>>(x, idx_wproj, wbuf, total);
  }
  // 10) indexer scores + causal
  mla_idx_score<<<dim3(S_LEN / 16, S_LEN / 64), 128, 0, stream>>>(iq_bf, ik_bf, wbuf, iscore);
  // 11) top-k threshold per query row
  mla_topk_thresh<<<S_LEN, 256, 0, stream>>>(iscore, thresh, TOPK);
  // 12) sparse flash attention (async K/V staging)
  mla_flash_attn<<<dim3(S_LEN / 128, NHEAD), 256, 0, stream>>>(q_bf, k_bf, vt_bf, iscore,
                                                               thresh, obuf);
  // 13) out = o @ wo^T
  mla_gemm_bt<<<dim3(S_LEN / GBM, DIMN / GBN), 256, 0, stream>>>(obuf, wo, out, S_LEN, DIMN,
                                                                 NHEAD * VDIM);
}
